// MoMFormer_5437428597522
// MI455X (gfx1250) — matrix operand order repaired and checked by reading, not compiled, not run on hardware
//
#include <hip/hip_runtime.h>
#include <hip/hip_bf16.h>
#include <hip/hip_fp16.h>

typedef __attribute__((ext_vector_type(16))) _Float16 v16h;
typedef __attribute__((ext_vector_type(8)))  _Float16 v8h;
typedef __attribute__((ext_vector_type(4)))  _Float16 v4h;
typedef __attribute__((ext_vector_type(8)))  float    v8f;

#define WMMA_F16(A,B,C) __builtin_amdgcn_wmma_f32_16x16x32_f16(false,(A),false,(B),(short)0,(C),false,false)

// BN eval-mode scale with running_var=1: 1/sqrt(1+1e-5)
#define BN_RSQRT 0.9999950000374997f

// ---- WMMA fragment loaders (wave32, 16x16x32 f16) ------------------------
// A matrix 16x32: lane L (L&15)=M row; kb=(L&16)?8:0; elems 0..7 -> K kb..kb+7,
// elems 8..15 -> K kb+16..kb+23.  Source: row-major [M][K] in LDS.
__device__ __forceinline__ v16h ld_a_frag(const _Float16* base, int stride,
                                          int mbase, int kofs, int lane) {
  int m  = mbase + (lane & 15);
  int kb = (lane & 16) ? 8 : 0;
  const _Float16* p = base + m * stride + kofs + kb;
  v8h lo = *(const v8h*)(p);
  v8h hi = *(const v8h*)(p + 16);
  v16h r;
#pragma unroll
  for (int i = 0; i < 8; ++i) { r[i] = lo[i]; r[i + 8] = hi[i]; }
  return r;
}

// B matrix 32x16: lane L (L&15)=N col; ko=(L&16)?8:0; elems 0..7 -> K ko..ko+7,
// elems 8..15 -> K ko+16..ko+23.
// Source: N-major (transposed) LDS tile [N][K] -> two contiguous 8-half runs per lane.
__device__ __forceinline__ v16h ld_bt_frag(const _Float16* base, int stride,
                                           int nbase, int kofs, int lane) {
  int n  = nbase + (lane & 15);
  int ko = (lane & 16) ? 8 : 0;
  const _Float16* p = base + n * stride + kofs + ko;
  v8h lo = *(const v8h*)(p);
  v8h hi = *(const v8h*)(p + 16);
  v16h r;
#pragma unroll
  for (int i = 0; i < 8; ++i) { r[i] = lo[i]; r[i + 8] = hi[i]; }
  return r;
}

// ---- Kernel 1: QKV 1x1 conv as WMMA GEMM ---------------------------------
// Out[oc, p] = sum_k W[oc,k] * X[k,p]; scatter f16 result into window-major
// layout [qkv][b][hh][ww][head][t(64)][d(32)] so attention reads 4KB blocks.
__global__ __launch_bounds__(256) void qkv_gemm_kernel(
    const float* __restrict__ x, const float* __restrict__ qkv_w,
    const float* __restrict__ qkv_b, _Float16* __restrict__ qkvh) {
  __shared__ alignas(16) _Float16 Ah[64 * 32];
  __shared__ alignas(16) _Float16 Bt[64 * 32];
  const int tid = threadIdx.x, lane = tid & 31, wv = tid >> 5;
  const int wm = wv >> 1, wn = wv & 1;
  const int pbase = blockIdx.x * 64;
  const int ocb = blockIdx.y * 64;
  const int b = blockIdx.z;
  const float* xb = x + (size_t)b * 256 * 16384;

  v8f acc0 = {}; v8f acc1 = {};
  for (int k0 = 0; k0 < 256; k0 += 32) {
    { // stage weight tile [64 oc][32 k]
      int row = tid >> 2, kk = (tid & 3) * 8;
      const float* wp = qkv_w + (size_t)(ocb + row) * 256 + k0 + kk;
#pragma unroll
      for (int j = 0; j < 8; ++j) Ah[row * 32 + kk + j] = (_Float16)wp[j];
    }
    { // stage activation tile transposed: Bt[n][kk] = X[k0+kk][pbase+n]
      int n = tid & 63, kk0 = tid >> 6;
#pragma unroll
      for (int j = 0; j < 8; ++j) {
        int kk = kk0 + j * 4;
        Bt[n * 32 + kk] = (_Float16)xb[(size_t)(k0 + kk) * 16384 + pbase + n];
      }
    }
    if (k0 + 32 < 256)  // global_prefetch_b8 next K-slab
      __builtin_prefetch(&xb[(size_t)(k0 + 32 + (tid >> 6)) * 16384 + pbase + (tid & 63)], 0, 0);
    __syncthreads();
    v16h a  = ld_a_frag(Ah, 32, wm * 16, 0, lane);
    v16h b0 = ld_bt_frag(Bt, 32, wn * 32 + 0, 0, lane);
    v16h b1 = ld_bt_frag(Bt, 32, wn * 32 + 16, 0, lane);
    acc0 = WMMA_F16(a, b0, acc0);
    acc1 = WMMA_F16(a, b1, acc1);
    __syncthreads();
  }
  const int hi8 = (lane & 16) ? 8 : 0;
  const int nl = lane & 15;
#pragma unroll
  for (int nt = 0; nt < 2; ++nt) {
    v8f acc = nt ? acc1 : acc0;
    int p = pbase + wn * 32 + nt * 16 + nl;
    int h_ = p >> 7, w_ = p & 127;
    int hh = h_ >> 3, ww = w_ >> 3;
    int t = (h_ & 7) * 8 + (w_ & 7);
#pragma unroll
    for (int r = 0; r < 8; ++r) {
      int oc = ocb + wm * 16 + r + hi8;
      float v = acc[r] + qkv_b[oc];
      int qi = oc >> 8, c = oc & 255;
      int head = c >> 5, d = c & 31;
      size_t dst = ((size_t)(((qi * 8 + b) * 16 + hh) * 16 + ww) * 8 + head) * 2048
                   + (size_t)t * 32 + d;
      qkvh[dst] = (_Float16)v;
    }
  }
}

// ---- Kernel 2: window attention (64 tokens, hd=32) -----------------------
// One workgroup (4 waves) per (window, head). Q/K staged with CDNA5 async
// global->LDS DMA (ASYNCcnt); S=QK^T: K=32 -> 1 WMMA/tile; O=PV: 2 chained
// WMMAs per 16x16 output tile.
__global__ __launch_bounds__(128) void win_attn_kernel(
    const _Float16* __restrict__ qkvh, const float* __restrict__ rpb,
    float* __restrict__ attn_out) {
  __shared__ alignas(16) _Float16 Qs[64 * 32];
  __shared__ alignas(16) _Float16 Ks[64 * 32];
  __shared__ alignas(16) _Float16 Vt[32 * 64];  // transposed: [d][t]
  __shared__ alignas(16) float    Ss[64 * 64];
  __shared__ alignas(16) _Float16 Ps[64 * 64];
  __shared__ float red[128];

  const int tid = threadIdx.x, lane = tid & 31, wv = tid >> 5;
  const int head = blockIdx.x, win = blockIdx.y, b = blockIdx.z;
  const int hh = win >> 4, ww = win & 15;
  const size_t blk = ((size_t)((b * 16 + hh) * 16 + ww) * 8 + head) * 2048;
  const size_t sq = (size_t)8 * 16 * 16 * 8 * 2048;  // per-qkv stride
  const _Float16* Qg = qkvh + blk;
  const _Float16* Kg = qkvh + blk + sq;
  const _Float16* Vg = qkvh + blk + 2 * sq;

  { // stage Q,K via async DMA: 128 lanes x 16B = 2KB/issue, offset covers 2nd half
    unsigned q_lds = (unsigned)(uintptr_t)Qs + (unsigned)tid * 16u;
    unsigned k_lds = (unsigned)(uintptr_t)Ks + (unsigned)tid * 16u;
    unsigned long long qa = (unsigned long long)(uintptr_t)Qg + (unsigned long long)tid * 16u;
    unsigned long long ka = (unsigned long long)(uintptr_t)Kg + (unsigned long long)tid * 16u;
    asm volatile("global_load_async_to_lds_b128 %0, %1, off"
                 :: "v"(q_lds), "v"(qa) : "memory");
    asm volatile("global_load_async_to_lds_b128 %0, %1, off offset:2048"
                 :: "v"(q_lds), "v"(qa) : "memory");
    asm volatile("global_load_async_to_lds_b128 %0, %1, off"
                 :: "v"(k_lds), "v"(ka) : "memory");
    asm volatile("global_load_async_to_lds_b128 %0, %1, off offset:2048"
                 :: "v"(k_lds), "v"(ka) : "memory");
  }
  // stage V transposed (needs f16-granularity transpose -> VGPR path)
  for (int i = tid; i < 512; i += 128) {
    int t = i >> 3, d0 = (i & 7) * 4;
    v4h v = *(const v4h*)(Vg + t * 32 + d0);
#pragma unroll
    for (int j = 0; j < 4; ++j) Vt[(d0 + j) * 64 + t] = v[j];
  }
  asm volatile("s_wait_asynccnt 0" ::: "memory");
  __syncthreads();

  const int mb = wv * 16;
  const int hi8 = (lane & 16) ? 8 : 0;
  const int nl = lane & 15;
  { // S strip (16x64) = Q_strip @ K^T : 4 WMMAs, K=32 in one shot
    v16h a = ld_a_frag(Qs, 32, mb, 0, lane);
#pragma unroll
    for (int jt = 0; jt < 4; ++jt) {
      v16h bb = ld_bt_frag(Ks, 32, jt * 16, 0, lane);
      v8f z = {};
      v8f s = WMMA_F16(a, bb, z);
#pragma unroll
      for (int r = 0; r < 8; ++r)
        Ss[(mb + r + hi8) * 64 + jt * 16 + nl] = s[r];
    }
  }
  __syncthreads();

  // softmax: 2 threads per row (32 cols each), LDS pairwise reduction.
  // rel-pos bias idx = (dy+7)*15 + (dx+7)
  {
    const float scale = 0.17677669529663687f;  // 1/sqrt(32)
    const int row = tid >> 1, c0 = (tid & 1) * 32;
    const int yi = row >> 3, xi = row & 7;
    float mx = -1e30f;
    for (int j = 0; j < 32; ++j) {
      int col = c0 + j;
      int yj = col >> 3, xj = col & 7;
      int idx = (yi - yj + 7) * 15 + (xi - xj + 7);
      float s = Ss[row * 64 + col] * scale + rpb[idx * 8 + head];
      Ss[row * 64 + col] = s;
      mx = fmaxf(mx, s);
    }
    red[tid] = mx;
    __syncthreads();
    float m = fmaxf(red[row * 2], red[row * 2 + 1]);
    float sum = 0.f;
    for (int j = 0; j < 32; ++j) {
      float e = __expf(Ss[row * 64 + c0 + j] - m);
      Ss[row * 64 + c0 + j] = e;
      sum += e;
    }
    __syncthreads();
    red[tid] = sum;
    __syncthreads();
    float inv = 1.0f / (red[row * 2] + red[row * 2 + 1]);
    for (int j = 0; j < 32; ++j)
      Ps[row * 64 + c0 + j] = (_Float16)(Ss[row * 64 + c0 + j] * inv);
  }
  __syncthreads();

  { // O strip (16x32) = P_strip @ V : 2 tiles x 2 K-chunks = 4 WMMAs
    v16h a0 = ld_a_frag(Ps, 64, mb, 0, lane);
    v16h a1 = ld_a_frag(Ps, 64, mb, 32, lane);
#pragma unroll
    for (int nt = 0; nt < 2; ++nt) {
      v16h b0 = ld_bt_frag(Vt, 64, nt * 16, 0, lane);
      v16h b1 = ld_bt_frag(Vt, 64, nt * 16, 32, lane);
      v8f acc = {};
      acc = WMMA_F16(a0, b0, acc);
      acc = WMMA_F16(a1, b1, acc);
      int ch = head * 32 + nt * 16 + nl;
#pragma unroll
      for (int r = 0; r < 8; ++r) {
        int t = mb + r + hi8;
        int h_ = hh * 8 + (t >> 3), w_ = ww * 8 + (t & 7);
        attn_out[((size_t)(b * 256 + ch) * 128 + h_) * 128 + w_] = acc[r];
      }
    }
  }
}

// ---- Kernel 3: grouped 3x3 conv (groups=8) + BN + ReLU -------------------
__global__ __launch_bounds__(256) void local_conv_kernel(
    const float* __restrict__ x, const float* __restrict__ lw,
    const float* __restrict__ lb, const float* __restrict__ bng,
    const float* __restrict__ bnb, float* __restrict__ out) {
  __shared__ float wloc[288];
  const int tid = threadIdx.x;
  const int oc = blockIdx.y, b = blockIdx.z;
  const int ty = blockIdx.x >> 3, tx = blockIdx.x & 7;
  for (int i = tid; i < 288; i += 256) wloc[i] = lw[(size_t)oc * 288 + i];
  __syncthreads();
  const int group = oc >> 5;
  const float* xg = x + ((size_t)b * 256 + group * 32) * 16384;
  const int h = ty * 16 + (tid >> 4);
  const int w = tx * 16 + (tid & 15);
  float sum = lb[oc];
  for (int ic = 0; ic < 32; ++ic) {
    const float* xc = xg + (size_t)ic * 16384;
    const float* wk = wloc + ic * 9;
#pragma unroll
    for (int dh = 0; dh < 3; ++dh) {
      int hh_ = h + dh - 1;
      if ((unsigned)hh_ >= 128u) continue;
#pragma unroll
      for (int dw = 0; dw < 3; ++dw) {
        int ww_ = w + dw - 1;
        if ((unsigned)ww_ >= 128u) continue;
        sum += xc[hh_ * 128 + ww_] * wk[dh * 3 + dw];
      }
    }
  }
  float v = sum * (bng[oc] * BN_RSQRT) + bnb[oc];
  out[(((size_t)b * 256 + oc) * 128 + h) * 128 + w] = fmaxf(v, 0.f);
}

// ---- Kernel 4: asymmetric avg-pool fusion + local add --------------------
// ax: reflect-pad bottom by 1 then AvgPool(8,1) pad (3,0); ay analogous.
__global__ __launch_bounds__(256) void fuse_kernel(
    const float* __restrict__ attn, const float* __restrict__ local,
    float* __restrict__ fused) {
  size_t idx = (size_t)blockIdx.x * 256 + threadIdx.x;
  if (idx >= (size_t)8 * 256 * 16384) return;
  int w = (int)(idx & 127), h = (int)((idx >> 7) & 127);
  const float* base = attn + ((idx >> 14) << 14);
  float ax = 0.f, ay = 0.f;
#pragma unroll
  for (int k = 0; k < 8; ++k) {
    int i = h - 3 + k;
    if (i >= 0) { int ii = (i == 128) ? 126 : i; ax += base[ii * 128 + w]; }
    int j = w - 3 + k;
    if (j >= 0) { int jj = (j == 128) ? 126 : j; ay += base[h * 128 + jj]; }
  }
  fused[idx] = (ax + ay) * 0.125f + local[idx];
}

// ---- Kernel 5: reflect-pad(0,1,0,1) + depthwise 3x3 + BN -----------------
__global__ __launch_bounds__(256) void dw_conv_kernel(
    const float* __restrict__ fused, const float* __restrict__ dww,
    const float* __restrict__ dwb, const float* __restrict__ bng,
    const float* __restrict__ bnb, float* __restrict__ out) {
  size_t idx = (size_t)blockIdx.x * 256 + threadIdx.x;
  if (idx >= (size_t)8 * 256 * 16384) return;
  int w = (int)(idx & 127), h = (int)((idx >> 7) & 127);
  int c = (int)((idx >> 14) & 255);
  const float* base = fused + ((idx >> 14) << 14);
  float sum = dwb[c];
#pragma unroll
  for (int dh = 0; dh < 3; ++dh) {
    int i = h + dh - 1;
    if (i < 0) continue;
    int ii = (i == 128) ? 126 : i;  // reflect-appended row
#pragma unroll
    for (int dw_ = 0; dw_ < 3; ++dw_) {
      int j = w + dw_ - 1;
      if (j < 0) continue;
      int jj = (j == 128) ? 126 : j;
      sum += base[ii * 128 + jj] * dww[c * 9 + dh * 3 + dw_];
    }
  }
  out[idx] = sum * (bng[c] * BN_RSQRT) + bnb[c];
}

// ---- Kernel 6: pointwise 1x1 proj as WMMA GEMM ---------------------------
__global__ __launch_bounds__(256) void pw_gemm_kernel(
    const float* __restrict__ dwo, const float* __restrict__ pw_w,
    const float* __restrict__ pw_b, float* __restrict__ out) {
  __shared__ alignas(16) _Float16 Ah[64 * 32];
  __shared__ alignas(16) _Float16 Bt[64 * 32];
  const int tid = threadIdx.x, lane = tid & 31, wv = tid >> 5;
  const int wm = wv >> 1, wn = wv & 1;
  const int pbase = blockIdx.x * 64;
  const int ocb = blockIdx.y * 64;
  const int b = blockIdx.z;
  const float* db = dwo + (size_t)b * 256 * 16384;

  v8f acc0 = {}; v8f acc1 = {};
  for (int k0 = 0; k0 < 256; k0 += 32) {
    {
      int row = tid >> 2, kk = (tid & 3) * 8;
      const float* wp = pw_w + (size_t)(ocb + row) * 256 + k0 + kk;
#pragma unroll
      for (int j = 0; j < 8; ++j) Ah[row * 32 + kk + j] = (_Float16)wp[j];
    }
    {
      int n = tid & 63, kk0 = tid >> 6;
#pragma unroll
      for (int j = 0; j < 8; ++j) {
        int kk = kk0 + j * 4;
        Bt[n * 32 + kk] = (_Float16)db[(size_t)(k0 + kk) * 16384 + pbase + n];
      }
    }
    if (k0 + 32 < 256)
      __builtin_prefetch(&db[(size_t)(k0 + 32 + (tid >> 6)) * 16384 + pbase + (tid & 63)], 0, 0);
    __syncthreads();
    v16h a  = ld_a_frag(Ah, 32, wm * 16, 0, lane);
    v16h b0 = ld_bt_frag(Bt, 32, wn * 32 + 0, 0, lane);
    v16h b1 = ld_bt_frag(Bt, 32, wn * 32 + 16, 0, lane);
    acc0 = WMMA_F16(a, b0, acc0);
    acc1 = WMMA_F16(a, b1, acc1);
    __syncthreads();
  }
  const int hi8 = (lane & 16) ? 8 : 0;
  const int nl = lane & 15;
#pragma unroll
  for (int nt = 0; nt < 2; ++nt) {
    v8f acc = nt ? acc1 : acc0;
    int p = pbase + wn * 32 + nt * 16 + nl;
#pragma unroll
    for (int r = 0; r < 8; ++r) {
      int oc = ocb + wm * 16 + r + hi8;
      out[((size_t)b * 256 + oc) * 16384 + p] = acc[r] + pw_b[oc];
    }
  }
}

// ---- launch ---------------------------------------------------------------
extern "C" void kernel_launch(void* const* d_in, const int* in_sizes, int n_in,
                              void* d_out, int out_size, void* d_ws, size_t ws_size,
                              hipStream_t stream) {
  const float* x       = (const float*)d_in[0];
  const float* qkv_w   = (const float*)d_in[1];
  const float* qkv_b   = (const float*)d_in[2];
  const float* local_w = (const float*)d_in[3];
  const float* local_b = (const float*)d_in[4];
  const float* lbn_g   = (const float*)d_in[5];
  const float* lbn_b   = (const float*)d_in[6];
  const float* rpb     = (const float*)d_in[7];
  const float* dw_w    = (const float*)d_in[8];
  const float* dw_b    = (const float*)d_in[9];
  const float* pbn_g   = (const float*)d_in[10];
  const float* pbn_b   = (const float*)d_in[11];
  const float* pw_w    = (const float*)d_in[12];
  const float* pw_b    = (const float*)d_in[13];

  char* ws = (char*)d_ws;
  const size_t QKV_BYTES  = (size_t)3 * 8 * 256 * 8 * 2048 * 2;  // 192 MB f16
  const size_t PLANE      = (size_t)8 * 256 * 16384 * 4;         // 128 MB f32
  _Float16* qkvh  = (_Float16*)ws;
  float*    attn  = (float*)(ws + QKV_BYTES);
  float*    local = (float*)(ws + QKV_BYTES + PLANE);
  float*    fused = (float*)ws;                 // reuse qkv region (after attn)
  float*    dwo   = (float*)(ws + QKV_BYTES);   // reuse attn region (after fuse)
  float*    out   = (float*)d_out;

  const size_t total = (size_t)8 * 256 * 16384;
  const int eb = (int)((total + 255) / 256);

  qkv_gemm_kernel<<<dim3(256, 12, 8), 256, 0, stream>>>(x, qkv_w, qkv_b, qkvh);
  win_attn_kernel<<<dim3(8, 256, 8), 128, 0, stream>>>(qkvh, rpb, attn);
  local_conv_kernel<<<dim3(64, 256, 8), 256, 0, stream>>>(x, local_w, local_b,
                                                          lbn_g, lbn_b, local);
  fuse_kernel<<<eb, 256, 0, stream>>>(attn, local, fused);
  dw_conv_kernel<<<eb, 256, 0, stream>>>(fused, dw_w, dw_b, pbn_g, pbn_b, dwo);
  pw_gemm_kernel<<<dim3(256, 4, 8), 256, 0, stream>>>(dwo, pw_w, pw_b, out);
}
